// Model_86071144611864
// MI455X (gfx1250) — matrix operand order repaired and checked by reading, not compiled, not run on hardware
//
#include <hip/hip_runtime.h>
#include <hip/hip_bf16.h>

#define H 128
#define NE 300000
#define NL 200000
#define TOTAL_ROWS 41000
#define MAX_DST 20000

typedef __attribute__((ext_vector_type(16))) __bf16 v16bf;
typedef __attribute__((ext_vector_type(8)))  __bf16 v8bf;
typedef __attribute__((ext_vector_type(8)))  float  v8f;

// node types: 0=drug(4000) 1=adr(6000) 2=gene(20000) 3=disease(8000) 4=dp(3000)
static const int NT_N[5]   = {4000, 6000, 20000, 8000, 3000};
static const int NT_OFF[5] = {0, 4000, 10000, 30000, 38000};
// edge types in reference order
static const int ET_SRC[9] = {0, 2, 3, 2, 2, 1, 0, 4, 3};
static const int ET_DST[9] = {1, 0, 4, 3, 2, 0, 2, 3, 2};

// ---------------- elementwise helpers ----------------

__global__ void zero_f32(float* __restrict__ p, int n4) {
    int i = blockIdx.x * blockDim.x + threadIdx.x;
    if (i < n4) ((float4*)p)[i] = make_float4(0.f, 0.f, 0.f, 0.f);
}

__global__ void relu_f32(float* __restrict__ p, int n4) {
    int i = blockIdx.x * blockDim.x + threadIdx.x;
    if (i < n4) {
        float4 v = ((float4*)p)[i];
        v.x = fmaxf(v.x, 0.f); v.y = fmaxf(v.y, 0.f);
        v.z = fmaxf(v.z, 0.f); v.w = fmaxf(v.w, 0.f);
        ((float4*)p)[i] = v;
    }
}

// x[i,:] = emb[idx[i],:]   (32 lanes per row, 4 floats per lane)
__global__ void embed_gather(const float* __restrict__ emb, const int* __restrict__ idx,
                             float* __restrict__ out, int n) {
    int tid = blockIdx.x * blockDim.x + threadIdx.x;
    if (tid >= n * 32) return;
    int i = tid >> 5;
    int j = (tid & 31) * 4;
    int s = idx[i];
    *(float4*)(out + (long)i * H + j) = *(const float4*)(emb + (long)s * H + j);
}

// agg[col[e],:] += x_src[row[e],:]   (32 lanes per edge, 4 floats per lane)
__global__ void scatter_add(const float* __restrict__ xsrc,
                            const int* __restrict__ row, const int* __restrict__ col,
                            float* __restrict__ agg, int ne) {
    long tid = (long)blockIdx.x * blockDim.x + threadIdx.x;
    if (tid >= (long)ne * 32) return;
    int e = (int)(tid >> 5);
    int j = (int)(tid & 31) * 4;
    int r = row[e];
    int c = col[e];
    if (e + 8 < ne) { // warm L2/WGP$ for an upcoming edge's source row
        __builtin_prefetch(xsrc + (long)row[e + 8] * H + j, 0, 1);
    }
    float4 v = *(const float4*)(xsrc + (long)r * H + j);
    float* d = agg + (long)c * H + j;
    atomicAdd(d + 0, v.x);
    atomicAdd(d + 1, v.y);
    atomicAdd(d + 2, v.z);
    atomicAdd(d + 3, v.w);
}

// ---------------- weight pre-pack into WMMA B-fragment order ----------------
// B frag (16x16x32 bf16): lane N = lane&15; this code packs VGPR v with K = 16*(lane>>4) + 2v, +1 (same contiguous order as A).
// Packed layout: [kcId(4)][wave(8)][lane(32)][16 bf16]  -> 32 contiguous bytes/lane.
__global__ void __launch_bounds__(256)
pack_weight_bf16(const float* __restrict__ W, __bf16* __restrict__ out) {
    int t = blockIdx.x * blockDim.x + threadIdx.x;   // 0..1023
    int lane = t & 31;
    int wave = (t >> 5) & 7;
    int kcId = t >> 8;                                // 0..3
    int mn = lane & 15, half = lane >> 4;
    int col = wave * 16 + mn;
    __bf16* o = out + (long)t * 16;
    #pragma unroll
    for (int v = 0; v < 8; ++v) {
        int k = kcId * 32 + half * 16 + 2 * v;
        o[2 * v]     = (__bf16)W[(long)k * H + col];
        o[2 * v + 1] = (__bf16)W[(long)(k + 1) * H + col];
    }
}

// ---------------- fused per-edge-type GraphConv GEMM ----------------
// out[r,:] += agg[r,:] @ rel_w + rel_b + xdst[r,:] @ root_w
// block = 256 threads (8 wave32); block owns a 16-row tile, each wave a 16-col tile.
// A tiles staged as bf16 row-major in LDS; B loaded pre-packed (2x b128 per frag).
__global__ void __launch_bounds__(256)
hetero_gemm(const float* __restrict__ agg, const float* __restrict__ xdst,
            const __bf16* __restrict__ pw_rel, const float* __restrict__ rel_b,
            const __bf16* __restrict__ pw_root,
            float* __restrict__ out, int n_dst) {
    __shared__ __bf16 sA[16 * H]; // agg tile, bf16 (4 KB)
    __shared__ __bf16 sX[16 * H]; // root tile, bf16 (4 KB)
    const int row0 = blockIdx.x * 16;
    const int t = threadIdx.x;

    // cooperative stage + f32->bf16 convert (pairs -> v_cvt_pk_bf16_f32 + ds_store_b32)
    #pragma unroll
    for (int i = t; i < 16 * (H / 2); i += 256) {
        int r  = i >> 6;
        int c2 = (i & 63) * 2;
        int gr = row0 + r;
        float2 va = make_float2(0.f, 0.f), vx = va;
        if (gr < n_dst) {
            va = *(const float2*)(agg  + (long)gr * H + c2);
            vx = *(const float2*)(xdst + (long)gr * H + c2);
        }
        sA[r * H + c2]     = (__bf16)va.x;
        sA[r * H + c2 + 1] = (__bf16)va.y;
        sX[r * H + c2]     = (__bf16)vx.x;
        sX[r * H + c2 + 1] = (__bf16)vx.y;
    }
    __syncthreads();

    const int wave = t >> 5, lane = t & 31;
    const int col0 = wave * 16, mn = lane & 15, half = lane >> 4;

    v8f acc = {};
    // pass 0: agg @ rel_w ; pass 1: x_dst @ root_w
    #pragma unroll
    for (int pass = 0; pass < 2; ++pass) {
        const __bf16* As = pass ? sX : sA;
        const __bf16* Wp = pass ? pw_root : pw_rel;
        #pragma unroll
        for (int kcId = 0; kcId < 4; ++kcId) {
            // A frag: lane M = mn; elems 0..15 = K kc+half*16..kc+half*16+15
            const __bf16* ab = As + mn * H + kcId * 32 + half * 16;
            v8bf a_lo = *(const v8bf*)(ab);
            v8bf a_hi = *(const v8bf*)(ab + 8);
            // B frag: pre-packed, 32 contiguous bytes per lane
            const __bf16* bb = Wp + ((long)((kcId * 8 + wave) * 32 + lane)) * 16;
            v8bf b_lo = *(const v8bf*)(bb);
            v8bf b_hi = *(const v8bf*)(bb + 8);
            v16bf a = __builtin_shufflevector(a_lo, a_hi,
                        0,1,2,3,4,5,6,7,8,9,10,11,12,13,14,15);
            v16bf b = __builtin_shufflevector(b_lo, b_hi,
                        0,1,2,3,4,5,6,7,8,9,10,11,12,13,14,15);
            acc = __builtin_amdgcn_wmma_f32_16x16x32_bf16(
                /*neg_a=*/false, a, /*neg_b=*/false, b,
                /*c_mod=*/(short)0, acc, /*reuse_a=*/false, /*reuse_b=*/false);
        }
    }

    // epilogue: D layout -> VGPR r: M = r + 8*half, N = mn
    const float bias = rel_b[col0 + mn];
    #pragma unroll
    for (int r = 0; r < 8; ++r) {
        int grow = row0 + r + half * 8;
        if (grow < n_dst) {
            long idx = (long)grow * H + col0 + mn;
            out[idx] += acc[r] + bias;
        }
    }
}

// ---------------- cosine decoder: one wave32 per label ----------------
__global__ void cosine_decoder(const float* __restrict__ zdrug, const float* __restrict__ zadr,
                               const int* __restrict__ ia, const int* __restrict__ ib,
                               float* __restrict__ out, int nl) {
    int wid  = (blockIdx.x * blockDim.x + threadIdx.x) >> 5;
    int lane = threadIdx.x & 31;
    if (wid >= nl) return;
    const float4 a = *(const float4*)(zdrug + (long)ia[wid] * H + lane * 4);
    const float4 b = *(const float4*)(zadr  + (long)ib[wid] * H + lane * 4);
    float dot = a.x * b.x + a.y * b.y + a.z * b.z + a.w * b.w;
    float na  = a.x * a.x + a.y * a.y + a.z * a.z + a.w * a.w;
    float nb  = b.x * b.x + b.y * b.y + b.z * b.z + b.w * b.w;
    #pragma unroll
    for (int off = 16; off > 0; off >>= 1) {
        dot += __shfl_xor(dot, off, 32);
        na  += __shfl_xor(na,  off, 32);
        nb  += __shfl_xor(nb,  off, 32);
    }
    if (lane == 0)
        out[wid] = dot / fmaxf(sqrtf(na) * sqrtf(nb), 1e-6f);
}

// ---------------- driver ----------------
extern "C" void kernel_launch(void* const* d_in, const int* in_sizes, int n_in,
                              void* d_out, int out_size, void* d_ws, size_t ws_size,
                              hipStream_t stream) {
    (void)in_sizes; (void)n_in; (void)out_size; (void)ws_size;

    // input layout (setup_inputs dict order):
    // [0..4]   x_dict int32 per node type
    // [5..13]  edge_index int32 [2,E] per edge type (row at +0, col at +E)
    // [14]     edge_label_index int32 [2,NL]
    // [15..19] emb_tables f32 per node type
    // [20...]  params: for l in 0..2, for et in 0..8: rel_w, rel_b, root_w
    const int* x_idx[5];
    const float* emb[5];
    for (int nt = 0; nt < 5; ++nt) {
        x_idx[nt] = (const int*)d_in[nt];
        emb[nt]   = (const float*)d_in[15 + nt];
    }
    const int* eidx[9];
    for (int et = 0; et < 9; ++et) eidx[et] = (const int*)d_in[5 + et];
    const int* eli = (const int*)d_in[14];

    float* bufA = (float*)d_ws;                  // [41000,128] f32
    float* bufB = bufA + (long)TOTAL_ROWS * H;   // [41000,128] f32
    float* agg  = bufB + (long)TOTAL_ROWS * H;   // [20000,128] f32
    __bf16* pw_rel  = (__bf16*)(agg + (long)MAX_DST * H); // [128,128] bf16 packed
    __bf16* pw_root = pw_rel + H * H;                     // [128,128] bf16 packed

    float* x_cur  = bufA;
    float* x_next = bufB;

    // embedding lookup (indices are identity ids, but do the gather faithfully)
    for (int nt = 0; nt < 5; ++nt) {
        int n = NT_N[nt];
        int threads = n * 32;
        embed_gather<<<(threads + 255) / 256, 256, 0, stream>>>(
            emb[nt], x_idx[nt], x_cur + (long)NT_OFF[nt] * H, n);
    }

    const int tot4 = TOTAL_ROWS * H / 4;
    for (int l = 0; l < 3; ++l) {
        zero_f32<<<(tot4 + 255) / 256, 256, 0, stream>>>(x_next, tot4);
        for (int et = 0; et < 9; ++et) {
            int src = ET_SRC[et], dst = ET_DST[et];
            int ndst = NT_N[dst];
            int a4 = ndst * H / 4;
            zero_f32<<<(a4 + 255) / 256, 256, 0, stream>>>(agg, a4);

            long sthreads = (long)NE * 32;
            scatter_add<<<(int)((sthreads + 255) / 256), 256, 0, stream>>>(
                x_cur + (long)NT_OFF[src] * H, eidx[et], eidx[et] + NE, agg, NE);

            const int pbase = 20 + (l * 9 + et) * 3;
            const float* rel_w  = (const float*)d_in[pbase + 0];
            const float* rel_b  = (const float*)d_in[pbase + 1];
            const float* root_w = (const float*)d_in[pbase + 2];

            pack_weight_bf16<<<4, 256, 0, stream>>>(rel_w,  pw_rel);
            pack_weight_bf16<<<4, 256, 0, stream>>>(root_w, pw_root);

            hetero_gemm<<<(ndst + 15) / 16, 256, 0, stream>>>(
                agg, x_cur + (long)NT_OFF[dst] * H, pw_rel, rel_b, pw_root,
                x_next + (long)NT_OFF[dst] * H, ndst);
        }
        if (l < 2)
            relu_f32<<<(tot4 + 255) / 256, 256, 0, stream>>>(x_next, tot4);
        float* tmp = x_cur; x_cur = x_next; x_next = tmp;
    }

    // decoder on z = x_cur
    long cthreads = (long)NL * 32;
    cosine_decoder<<<(int)((cthreads + 255) / 256), 256, 0, stream>>>(
        x_cur + (long)NT_OFF[0] * H, x_cur + (long)NT_OFF[1] * H,
        eli, eli + NL, (float*)d_out, NL);
}
